// AttentionHead_77455440216747
// MI455X (gfx1250) — matrix operand order repaired and checked by reading, not compiled, not run on hardware
//
#include <hip/hip_runtime.h>
#include <hip/hip_bf16.h>

typedef __attribute__((ext_vector_type(8)))  __bf16 v8bf;
typedef __attribute__((ext_vector_type(16))) __bf16 v16bf;
typedef __attribute__((ext_vector_type(8)))  float  v8f;

#define WMMA_BF16(A, B, C) \
    __builtin_amdgcn_wmma_f32_16x16x32_bf16(false, (A), false, (B), (short)0, (C), false, false)

#define CAT16(lo, hi) __builtin_shufflevector((lo), (hi), \
    0,1,2,3,4,5,6,7,8,9,10,11,12,13,14,15)

// float -> bf16 round-to-nearest-even via bit ops (cold paths only)
__device__ __forceinline__ __bf16 f2bf(float f) {
    union { float f; unsigned u; } in; in.f = f;
    unsigned r = (in.u + 0x7FFFu + ((in.u >> 16) & 1u)) >> 16;
    union { unsigned short s; __bf16 b; } out; out.s = (unsigned short)r;
    return out.b;
}

// ---------------------------------------------------------------------------
// Streaming f32 -> bf16 conversion (x). 8 elems/thread, b128 in/out.
// ---------------------------------------------------------------------------
__global__ __launch_bounds__(256)
void cvt_bf16_kernel(const float* __restrict__ in, __bf16* __restrict__ out, size_t n)
{
    size_t i = ((size_t)blockIdx.x * blockDim.x + threadIdx.x) * 8;
    if (i >= n) return;
    const float4* p = reinterpret_cast<const float4*>(in + i);
    float4 a = p[0], b = p[1];
    v8bf o;
    o[0]=f2bf(a.x); o[1]=f2bf(a.y); o[2]=f2bf(a.z); o[3]=f2bf(a.w);
    o[4]=f2bf(b.x); o[5]=f2bf(b.y); o[6]=f2bf(b.z); o[7]=f2bf(b.w);
    *reinterpret_cast<v8bf*>(out + i) = o;
}

// ---------------------------------------------------------------------------
// W[H,D] f32 -> Wt[D,H] bf16 (tiny: 64K elements per matrix)
// ---------------------------------------------------------------------------
__global__ __launch_bounds__(256)
void cvt_transpose_kernel(const float* __restrict__ in, __bf16* __restrict__ out,
                          int H, int D)
{
    int idx = blockIdx.x * blockDim.x + threadIdx.x;     // idx = h*D + d
    if (idx >= H * D) return;
    int h = idx / D, d = idx - h * D;
    out[(size_t)d * H + h] = f2bf(in[idx]);
}

// ---------------------------------------------------------------------------
// Stage 1: 16-row tile of (xb[M,H] @ W) -> bf16, using Wt[D,H].
// One wave per tile; all operands contiguous b128 loads; zero hot-loop cvt.
// transposed==0: out[M,D] row-major. transposed!=0: outT[B][D][T].
// ---------------------------------------------------------------------------
__global__ __launch_bounds__(32)
void proj_kernel(const __bf16* __restrict__ xb,
                 const __bf16* __restrict__ Wt,
                 __bf16* __restrict__ out,
                 int H, int D, int T, float oscale, int transposed)
{
    const int lane = threadIdx.x & 31;
    const int r    = lane & 15;      // A row / B,C column
    const int g    = lane >> 4;      // lane-half -> K sub-range
    const int kb8  = 8 * g;
    const int row0 = blockIdx.x * 16;

    v8f acc[4] = {};
    const __bf16* xr = xb + (size_t)(row0 + r) * H;

    for (int h0 = 0; h0 < H; h0 += 32) {
        if (h0 + 32 < H) __builtin_prefetch(xr + h0 + 32, 0, 3);
        // A tile 16x32: lane r holds K = {kb8..kb8+7, 16+kb8..16+kb8+7}
        v8bf a0 = *reinterpret_cast<const v8bf*>(xr + h0 + kb8);
        v8bf a1 = *reinterpret_cast<const v8bf*>(xr + h0 + 16 + kb8);
        v16bf a = CAT16(a0, a1);
        #pragma unroll
        for (int t = 0; t < 4; ++t) {
            // B tile 32x16: lane col = t*16+r, K = {kb8..kb8+7, 16+kb8..16+kb8+7}
            const __bf16* wr = Wt + (size_t)(t * 16 + r) * H + h0 + kb8;
            v8bf b0 = reinterpret_cast<const v8bf*>(wr)[0];
            v8bf b1 = reinterpret_cast<const v8bf*>(wr + 16)[0];
            acc[t] = WMMA_BF16(a, CAT16(b0, b1), acc[t]);
        }
    }

    // epilogue: C layout (lane = col, vgpr i + 8g = row), store bf16
    if (!transposed) {
        #pragma unroll
        for (int t = 0; t < 4; ++t)
            #pragma unroll
            for (int i = 0; i < 8; ++i)
                out[(size_t)(row0 + i + 8 * g) * D + t * 16 + r] =
                    f2bf(acc[t][i] * oscale);
    } else {
        const int bidx = row0 / T;               // tile never straddles a batch
        const int tt0  = row0 - bidx * T;
        #pragma unroll
        for (int t = 0; t < 4; ++t)
            #pragma unroll
            for (int i = 0; i < 8; ++i)
                out[((size_t)bidx * D + t * 16 + r) * (size_t)T + tt0 + i + 8 * g] =
                    f2bf(acc[t][i] * oscale);
    }
}

// ---------------------------------------------------------------------------
// Stage 2: causal flash attention, bf16 Q[B,T,D] (pre-scaled by 1/sqrt(D)),
// bf16 K[B,T,D], bf16 Vt[B,D,T]. One wave per (batch, 16-query tile),
// 32 keys/step: 4 WMMA for S, online softmax, P via LDS bf16, 4 WMMA for PV.
// ---------------------------------------------------------------------------
__global__ __launch_bounds__(32)
void flash_attn_kernel(const __bf16* __restrict__ Q,
                       const __bf16* __restrict__ K,
                       const __bf16* __restrict__ Vt,
                       float* __restrict__ O,
                       int T, int D)
{
    __shared__ __bf16 sP[16][32];    // P tile, row-major bf16 (64B rows, aligned)

    const int lane = threadIdx.x & 31;
    const int r    = lane & 15;
    const int g    = lane >> 4;
    const int kb8  = 8 * g;
    const int b    = blockIdx.y;
    const int q0   = blockIdx.x * 16;

    const __bf16* Qb = Q  + (size_t)b * T * D;
    const __bf16* Kb = K  + (size_t)b * T * D;
    const __bf16* Vb = Vt + (size_t)b * D * T;

    // Q tile in A layout: contiguous b128 pairs per 32-wide half of D
    v16bf qa[2];
    {
        const __bf16* qr = Qb + (size_t)(q0 + r) * D;
        #pragma unroll
        for (int h = 0; h < 2; ++h) {
            v8bf g0 = *reinterpret_cast<const v8bf*>(qr + h * 32 + kb8);
            v8bf g1 = *reinterpret_cast<const v8bf*>(qr + h * 32 + 16 + kb8);
            qa[h] = CAT16(g0, g1);
        }
    }

    v8f acc[4] = {};
    float m_i[8], l_i[8];
    #pragma unroll
    for (int i = 0; i < 8; ++i) { m_i[i] = -__builtin_inff(); l_i[i] = 0.f; }

    const int kt_end = (q0 + 15) / 32;           // inclusive causal bound
    for (int kt = 0; kt <= kt_end; ++kt) {
        const int kbase = kt * 32;

        // ---- S = Q K^T : two 16x16 f32 C tiles; K rows are contiguous b128s
        v8f S[2] = {};
        #pragma unroll
        for (int st = 0; st < 2; ++st) {
            const __bf16* kr = Kb + (size_t)(kbase + st * 16 + r) * D + kb8;
            #pragma unroll
            for (int h = 0; h < 2; ++h) {
                v8bf b0 = reinterpret_cast<const v8bf*>(kr + h * 32)[0];
                v8bf b1 = reinterpret_cast<const v8bf*>(kr + h * 32 + 16)[0];
                S[st] = WMMA_BF16(qa[h], CAT16(b0, b1), S[st]);
            }
        }

        // ---- causal mask + per-step row max
        float tmax[8];
        #pragma unroll
        for (int i = 0; i < 8; ++i) {
            const int qrow = q0 + i + 8 * g;
            float s0 = (kbase + r      <= qrow) ? S[0][i] : -__builtin_inff();
            float s1 = (kbase + 16 + r <= qrow) ? S[1][i] : -__builtin_inff();
            S[0][i] = s0; S[1][i] = s1;
            tmax[i] = fmaxf(s0, s1);
        }
        #pragma unroll
        for (int m = 1; m < 16; m <<= 1)         // reduce within 16-lane halves
            #pragma unroll
            for (int i = 0; i < 8; ++i)
                tmax[i] = fmaxf(tmax[i], __shfl_xor(tmax[i], m, 32));

        // ---- online softmax; P -> LDS as bf16 (C layout scatter)
        #pragma unroll
        for (int i = 0; i < 8; ++i) {
            const float mn = fmaxf(m_i[i], tmax[i]);
            const float sc = __expf(m_i[i] - mn);
            const float p0 = __expf(S[0][i] - mn);
            const float p1 = __expf(S[1][i] - mn);
            sP[i + 8 * g][r]      = f2bf(p0);
            sP[i + 8 * g][16 + r] = f2bf(p1);
            float rs = p0 + p1;
            #pragma unroll
            for (int m = 1; m < 16; m <<= 1)
                rs += __shfl_xor(rs, m, 32);
            l_i[i] = l_i[i] * sc + rs;
            m_i[i] = mn;
            #pragma unroll
            for (int t = 0; t < 4; ++t) acc[t][i] *= sc;
        }
        __syncthreads();

        // ---- P in A layout: two aligned ds b128 loads
        v8bf p0 = *reinterpret_cast<const v8bf*>(&sP[r][kb8]);
        v8bf p1 = *reinterpret_cast<const v8bf*>(&sP[r][16 + kb8]);
        v16bf pa = CAT16(p0, p1);
        __syncthreads();

        // ---- O += P @ V : Vt rows contiguous per lane's d column
        #pragma unroll
        for (int t = 0; t < 4; ++t) {
            const __bf16* vr = Vb + (size_t)(t * 16 + r) * T + kbase + kb8;
            v8bf b0 = reinterpret_cast<const v8bf*>(vr)[0];
            v8bf b1 = reinterpret_cast<const v8bf*>(vr + 16)[0];
            acc[t] = WMMA_BF16(pa, CAT16(b0, b1), acc[t]);
        }
    }

    // ---- epilogue: O / l, f32 out
    float* Ob = O + (size_t)b * T * D;
    #pragma unroll
    for (int i = 0; i < 8; ++i) {
        const float inv = 1.f / l_i[i];
        const int row = q0 + i + 8 * g;
        #pragma unroll
        for (int t = 0; t < 4; ++t)
            Ob[(size_t)row * D + t * 16 + r] = acc[t][i] * inv;
    }
}

// ---------------------------------------------------------------------------
extern "C" void kernel_launch(void* const* d_in, const int* in_sizes, int n_in,
                              void* d_out, int out_size, void* d_ws, size_t ws_size,
                              hipStream_t stream)
{
    const float* x  = (const float*)d_in[0];
    const float* Wk = (const float*)d_in[1];
    const float* Wq = (const float*)d_in[2];
    const float* Wv = (const float*)d_in[3];

    const int B = 8, T = 2048, H = 1024, D = 64;
    const size_t NX = (size_t)B * T * H;     // 16,777,216
    const size_t NW = (size_t)H * D;         // 65,536
    const size_t NP = (size_t)B * T * D;     // 1,048,576

    __bf16* xb  = (__bf16*)d_ws;
    __bf16* wtq = xb  + NX;
    __bf16* wtk = wtq + NW;
    __bf16* wtv = wtk + NW;
    __bf16* qb  = wtv + NW;
    __bf16* kb  = qb  + NP;
    __bf16* vtb = kb  + NP;                  // [B][D][T]

    // conversions
    hipLaunchKernelGGL(cvt_bf16_kernel, dim3((unsigned)(NX / 8 / 256)), dim3(256),
                       0, stream, x, xb, NX);
    hipLaunchKernelGGL(cvt_transpose_kernel, dim3((unsigned)(NW / 256)), dim3(256),
                       0, stream, Wq, wtq, H, D);
    hipLaunchKernelGGL(cvt_transpose_kernel, dim3((unsigned)(NW / 256)), dim3(256),
                       0, stream, Wk, wtk, H, D);
    hipLaunchKernelGGL(cvt_transpose_kernel, dim3((unsigned)(NW / 256)), dim3(256),
                       0, stream, Wv, wtv, H, D);

    // projections (Q pre-scaled by 1/sqrt(D); V stored transposed [B,D,T])
    dim3 blk(32);
    dim3 g1((B * T) / 16);
    hipLaunchKernelGGL(proj_kernel, g1, blk, 0, stream, xb, wtq, qb,  H, D, T, 0.125f, 0);
    hipLaunchKernelGGL(proj_kernel, g1, blk, 0, stream, xb, wtk, kb,  H, D, T, 1.0f,   0);
    hipLaunchKernelGGL(proj_kernel, g1, blk, 0, stream, xb, wtv, vtb, H, D, T, 1.0f,   1);

    // fused causal attention
    dim3 g2(T / 16, B);
    hipLaunchKernelGGL(flash_attn_kernel, g2, blk, 0, stream, qb, kb, vtb,
                       (float*)d_out, T, D);
}
